// TrajGRUCell_3942779978021
// MI455X (gfx1250) — matrix operand order repaired and checked by reading, not compiled, not run on hardware
//
#include <hip/hip_runtime.h>
#include <hip/hip_bf16.h>
#include <math.h>

// ---------------------------------------------------------------------------
// TrajGRU cell for MI455X (gfx1250), wave32 + bf16 WMMA.
//   B=4, CIN=8, CH=64, H=W=128, L=5, K=3  -> NPIX = 65536 pixels
// Block tile: 128M x 64N (8 waves, each 16M x 64N = 4 v8f accumulators).
// B panels resident in LDS; activations halo-padded so the K loop is fully
// branch-free: unconditional global b128 loads at immediate offsets + WMMA.
// ---------------------------------------------------------------------------

typedef __attribute__((ext_vector_type(16))) __bf16 v16bf;
typedef __attribute__((ext_vector_type(8)))  __bf16 v8bf;
typedef __attribute__((ext_vector_type(8)))  float  v8f;

#define NB   4
#define NCIN 8
#define NCH  64
#define NH   128
#define NW   128
#define PH   (NH + 2)           // halo-padded height (130)
#define PW   (NW + 2)           // halo-padded width  (130)
#define NL   5
#define HW   (NH * NW)          // 16384
#define NPIX (NB * HW)          // 65536
#define N3   (3 * NCH)          // 192
#define KX   96                 // 72 padded (x-part of xh rows)
#define KH   576                // 9*64 = 18x32
#define KHH  288                // Wh K half (2 phases, 36KB LDS panel)
#define KW1  320                // 5*64 = 10x32
#define KF   864                // 9 taps * 96 padded channels = 27x32
#define CXH  96                 // channels-last row length of concat(x,h) padded

static __device__ inline v16bf join8(v8bf lo, v8bf hi) {
    union { v16bf v; v8bf h[2]; } u;
    u.h[0] = lo; u.h[1] = hi;
    return u.v;
}

// ---------------------------------------------------------------------------
// Repack kernels: halo-padded channels-last bf16 activations + [N][K] panels.
// ---------------------------------------------------------------------------
__global__ void repack_xh(const float* __restrict__ x, const float* __restrict__ h,
                          __bf16* __restrict__ xh) {
    int i = blockIdx.x * blockDim.x + threadIdx.x;     // NB*PH*PW*CXH
    if (i >= NB * PH * PW * CXH) return;
    int pp = i / CXH, ci = i % CXH;
    int b = pp / (PH * PW);
    int rr = pp - b * (PH * PW);
    int yy = rr / PW, xx = rr % PW;
    float v = 0.f;
    if (yy >= 1 && yy <= NH && xx >= 1 && xx <= NW) {
        int rem = (yy - 1) * NW + (xx - 1);
        if (ci < NCIN)     v = x[(b * NCIN + ci) * HW + rem];
        else if (ci < 72)  v = h[(b * NCH + (ci - NCIN)) * HW + rem];
    }
    xh[(size_t)pp * CXH + ci] = (__bf16)v;
}

__global__ void fill_zero_bf16(__bf16* __restrict__ p, int n) {
    int i = blockIdx.x * blockDim.x + threadIdx.x;
    if (i < n) p[i] = (__bf16)0.0f;
}

__global__ void repack_wf(const float* __restrict__ Wf, __bf16* __restrict__ Wt) {
    int i = blockIdx.x * blockDim.x + threadIdx.x;     // 16*864
    if (i >= 16 * KF) return;
    int n = i / KF, k = i % KF;
    int tap = k / CXH, ci = k % CXH;
    float v = 0.f;
    if (n < 10 && ci < 72) v = Wf[(n * 72 + ci) * 9 + tap];
    Wt[(size_t)n * KF + k] = (__bf16)v;
}

__global__ void repack_wx(const float* __restrict__ Wx, __bf16* __restrict__ Wt) {
    int i = blockIdx.x * blockDim.x + threadIdx.x;     // 192*96
    if (i >= N3 * KX) return;
    int n = i / KX, k = i % KX;
    float v = 0.f;
    if (k < 72) {
        int tap = k / NCIN, ci = k % NCIN;
        v = Wx[((n * NCIN + ci) * 3 + tap / 3) * 3 + tap % 3];
    }
    Wt[(size_t)n * KX + k] = (__bf16)v;
}

__global__ void repack_wh(const float* __restrict__ Wh, __bf16* __restrict__ Wt) {
    int i = blockIdx.x * blockDim.x + threadIdx.x;     // 192*576
    if (i >= N3 * KH) return;
    int n = i / KH, k = i % KH;
    int tap = k / NCH, ci = k % NCH;
    Wt[(size_t)n * KH + k] = (__bf16)Wh[((n * NCH + ci) * 3 + tap / 3) * 3 + tap % 3];
}

__global__ void repack_ww(const float* __restrict__ Ww, __bf16* __restrict__ Wt) {
    int i = blockIdx.x * blockDim.x + threadIdx.x;     // 64*320
    if (i >= NCH * KW1) return;
    Wt[i] = (__bf16)Ww[i];   // layout [n][l*64+c] already matches
}

// ---------------------------------------------------------------------------
// Stage 1: flow conv as implicit GEMM (WMMA). M=NPIX, N=16 (10 used), K=864.
// k = tap*96 + ci. Whole 16x864 B panel (27KB) in LDS; branch-free K loop:
// per-lane pixel base computed once, taps are immediate offsets.
// ---------------------------------------------------------------------------
__global__ void __launch_bounds__(256)
gemm_flow(const __bf16* __restrict__ xh, const __bf16* __restrict__ Wt,
          const float* __restrict__ bfv, float* __restrict__ flows) {
    __shared__ __attribute__((aligned(64))) __bf16 bpan[16 * KF];
    const int tid  = threadIdx.x;
    const int lane = tid & 31, wv = tid >> 5;
    const int half = lane >> 4, ln = lane & 15;
    const int m0   = blockIdx.x * 128 + wv * 16;
    const int row  = m0 + ln;
    const int b    = row >> 14, rem = row & (HW - 1);
    const int y    = rem >> 7, x0 = rem & (NW - 1);
    const __bf16* abase =
        xh + ((size_t)(b * PH + (y + 1)) * PW + (x0 + 1)) * CXH + half * 8;

    for (int c = tid; c < 16 * KF / 8; c += 256)
        ((v8bf*)bpan)[c] = ((const v8bf*)Wt)[c];
    __syncthreads();

    v8f acc = {};
#pragma unroll
    for (int kc = 0; kc < KF / 32; ++kc) {
        const int tap = kc / 3;
        const int off = ((tap / 3 - 1) * PW + (tap % 3 - 1)) * CXH + (kc % 3) * 32;
        v8bf a0 = *(const v8bf*)(abase + off);
        v8bf a1 = *(const v8bf*)(abase + off + 16);
        v16bf am = join8(a0, a1);
        v16bf bm = join8(*(const v8bf*)&bpan[ln * KF + kc * 32 + half * 8],
                         *(const v8bf*)&bpan[ln * KF + kc * 32 + 16 + half * 8]);
        acc = __builtin_amdgcn_wmma_f32_16x16x32_bf16(
            false, am, false, bm, (short)0, acc, false, false);
    }
    if (ln < 10) {
        float bv = bfv[ln];
#pragma unroll
        for (int v = 0; v < 8; ++v) {
            int m = m0 + v + half * 8;
            int mb = m >> 14, mrem = m & (HW - 1);
            flows[(mb * 10 + ln) * HW + mrem] = acc[v] + bv;
        }
    }
}

// ---------------------------------------------------------------------------
// Stage 2: bilinear warp (border clamp) -> h_cat channels-last bf16 [p][320]
// ---------------------------------------------------------------------------
__global__ void warp_kernel(const float* __restrict__ h, const float* __restrict__ flows,
                            __bf16* __restrict__ hcat) {
    int t = blockIdx.x * blockDim.x + threadIdx.x;     // NB*NL*HW
    if (t >= NB * NL * HW) return;
    int b = t / (NL * HW);
    int r = t - b * (NL * HW);
    int l = r / HW;
    int rem = r - l * HW;
    int y = rem >> 7, x = rem & (NW - 1);
    float fx = flows[(b * 10 + l * 2 + 0) * HW + rem];
    float fy = flows[(b * 10 + l * 2 + 1) * HW + rem];
    float gx = fminf(fmaxf((float)x + fx, 0.f), (float)(NW - 1));
    float gy = fminf(fmaxf((float)y + fy, 0.f), (float)(NH - 1));
    float x0f = floorf(gx), y0f = floorf(gy);
    float wx = gx - x0f, wy = gy - y0f;
    int xi0 = (int)x0f, yi0 = (int)y0f;
    int xi1 = xi0 + 1 < NW ? xi0 + 1 : NW - 1;
    int yi1 = yi0 + 1 < NH ? yi0 + 1 : NH - 1;
    float w00 = (1.f - wy) * (1.f - wx), w01 = (1.f - wy) * wx;
    float w10 = wy * (1.f - wx),         w11 = wy * wx;
    int i00 = yi0 * NW + xi0, i01 = yi0 * NW + xi1;
    int i10 = yi1 * NW + xi0, i11 = yi1 * NW + xi1;
    const float* hb = h + (size_t)b * NCH * HW;
    __bf16* dst = hcat + (size_t)(b * HW + rem) * KW1 + l * NCH;
#pragma unroll 4
    for (int c = 0; c < NCH; ++c) {
        const float* hc = hb + c * HW;
        float v = w00 * hc[i00] + w01 * hc[i01] + w10 * hc[i10] + w11 * hc[i11];
        dst[c] = (__bf16)v;
    }
}

// ---------------------------------------------------------------------------
// Stage 3: 1x1 conv  h_cat[NPIX,320] x Ww^T[320,64] + bw  (WMMA)
// Whole 64x320 B panel (40KB) in LDS; contiguous A rows, branch-free.
// Writes hagg (f32, dense) and haggcl (bf16, halo-padded interior).
// ---------------------------------------------------------------------------
__global__ void __launch_bounds__(256)
gemm_ww(const __bf16* __restrict__ hcat, const __bf16* __restrict__ Wt,
        const float* __restrict__ bw, float* __restrict__ hagg,
        __bf16* __restrict__ haggcl) {
    __shared__ __attribute__((aligned(64))) __bf16 bpan[64 * KW1];
    const int tid  = threadIdx.x;
    const int lane = tid & 31, wv = tid >> 5;
    const int half = lane >> 4, ln = lane & 15;
    const int m0   = blockIdx.x * 128 + wv * 16;
    const int row  = m0 + ln;
    const __bf16* arow = hcat + (size_t)row * KW1 + half * 8;

    for (int c = tid; c < 64 * KW1 / 8; c += 256)
        ((v8bf*)bpan)[c] = ((const v8bf*)Wt)[c];
    __syncthreads();

    v8f acc[4] = {{}, {}, {}, {}};
#pragma unroll
    for (int kc = 0; kc < KW1 / 32; ++kc) {
        v8bf a0 = *(const v8bf*)(arow + kc * 32);
        v8bf a1 = *(const v8bf*)(arow + kc * 32 + 16);
        v16bf am = join8(a0, a1);
#pragma unroll
        for (int j = 0; j < 4; ++j) {
            v16bf bm = join8(*(const v8bf*)&bpan[(j * 16 + ln) * KW1 + kc * 32 + half * 8],
                             *(const v8bf*)&bpan[(j * 16 + ln) * KW1 + kc * 32 + 16 + half * 8]);
            acc[j] = __builtin_amdgcn_wmma_f32_16x16x32_bf16(
                false, am, false, bm, (short)0, acc[j], false, false);
        }
    }
#pragma unroll
    for (int j = 0; j < 4; ++j) {
        int nc = j * 16 + ln;
        float bv = bw[nc];
#pragma unroll
        for (int v = 0; v < 8; ++v) {
            int m = m0 + v + half * 8;
            int mb = m >> 14, mrem = m & (HW - 1);
            int my = mrem >> 7, mx = mrem & (NW - 1);
            float val = acc[j][v] + bv;
            hagg[(size_t)m * NCH + nc] = val;
            haggcl[((size_t)(mb * PH + my + 1) * PW + mx + 1) * NCH + nc] = (__bf16)val;
        }
    }
}

// ---------------------------------------------------------------------------
// Stage 4a: x_conv = x (*) Wx, implicit GEMM K=96 (WMMA).
// A rows come from halo-padded xh (x occupies ci 0..7). Taps >= 9 have zero
// weights, so they are clamped to tap 8 and loaded unconditionally (A*0 == 0).
// ---------------------------------------------------------------------------
__global__ void __launch_bounds__(256)
gemm_wx(const __bf16* __restrict__ xh, const __bf16* __restrict__ Wt,
        const float* __restrict__ bx, float* __restrict__ xg) {
    __shared__ __attribute__((aligned(64))) __bf16 bpan[64 * KX];
    const int tid  = threadIdx.x;
    const int lane = tid & 31, wv = tid >> 5;
    const int half = lane >> 4, ln = lane & 15;
    const int m0   = blockIdx.x * 128 + wv * 16;
    const int n0   = blockIdx.y * 64;
    const int row  = m0 + ln;
    const int b    = row >> 14, rem = row & (HW - 1);
    const int y    = rem >> 7, x0 = rem & (NW - 1);
    const __bf16* abase =
        xh + ((size_t)(b * PH + (y + 1)) * PW + (x0 + 1)) * CXH;

    const __bf16* wsrc = Wt + (size_t)n0 * KX;          // contiguous 64xKX panel
    for (int c = tid; c < 64 * KX / 8; c += 256)
        ((v8bf*)bpan)[c] = ((const v8bf*)wsrc)[c];
    __syncthreads();

    v8f acc[4] = {{}, {}, {}, {}};
#pragma unroll
    for (int kc = 0; kc < KX / 32; ++kc) {
        int tapA = kc * 4 + half;           // half 0 -> {4kc,4kc+2}; 1 -> {+1,+3}
        int tapB = tapA + 2;
        tapA = tapA > 8 ? 8 : tapA;         // zero weights beyond tap 8
        tapB = tapB > 8 ? 8 : tapB;
        v8bf a0 = *(const v8bf*)(abase + ((tapA / 3 - 1) * PW + (tapA % 3 - 1)) * CXH);
        v8bf a1 = *(const v8bf*)(abase + ((tapB / 3 - 1) * PW + (tapB % 3 - 1)) * CXH);
        v16bf am = join8(a0, a1);
#pragma unroll
        for (int j = 0; j < 4; ++j) {
            v16bf bm = join8(*(const v8bf*)&bpan[(j * 16 + ln) * KX + kc * 32 + half * 8],
                             *(const v8bf*)&bpan[(j * 16 + ln) * KX + kc * 32 + 16 + half * 8]);
            acc[j] = __builtin_amdgcn_wmma_f32_16x16x32_bf16(
                false, am, false, bm, (short)0, acc[j], false, false);
        }
    }
#pragma unroll
    for (int j = 0; j < 4; ++j) {
        int nc = n0 + j * 16 + ln;
        float bv = bx[nc];
#pragma unroll
        for (int v = 0; v < 8; ++v) {
            int m = m0 + v + half * 8;
            xg[(size_t)m * N3 + nc] = acc[j][v] + bv;
        }
    }
}

// ---------------------------------------------------------------------------
// Stage 4b: h_conv = h_agg (*) Wh, implicit GEMM K=576 (WMMA).
// B panel staged in two 64x288 phases (36KB LDS); acc persists across phases.
// A loads unconditional from halo-padded haggcl, taps are immediate offsets.
// ---------------------------------------------------------------------------
__global__ void __launch_bounds__(256)
gemm_wh(const __bf16* __restrict__ haggcl, const __bf16* __restrict__ Wt,
        const float* __restrict__ bh, float* __restrict__ hg) {
    __shared__ __attribute__((aligned(64))) __bf16 bpan[64 * KHH];
    const int tid  = threadIdx.x;
    const int lane = tid & 31, wv = tid >> 5;
    const int half = lane >> 4, ln = lane & 15;
    const int m0   = blockIdx.x * 128 + wv * 16;
    const int n0   = blockIdx.y * 64;
    const int row  = m0 + ln;
    const int b    = row >> 14, rem = row & (HW - 1);
    const int y    = rem >> 7, x0 = rem & (NW - 1);
    const __bf16* abase =
        haggcl + ((size_t)(b * PH + (y + 1)) * PW + (x0 + 1)) * NCH + half * 8;

    v8f acc[4] = {{}, {}, {}, {}};
#pragma unroll
    for (int ph = 0; ph < 2; ++ph) {
        // stage 64 x 288 slice of the weight panel
        for (int c = tid; c < 64 * (KHH / 8); c += 256) {
            int r  = c / (KHH / 8);
            int c8 = c % (KHH / 8);
            ((v8bf*)bpan)[r * (KHH / 8) + c8] =
                *(const v8bf*)(Wt + (size_t)(n0 + r) * KH + ph * KHH + c8 * 8);
        }
        __syncthreads();
#pragma unroll
        for (int kc2 = 0; kc2 < KHH / 32; ++kc2) {
            const int kc  = ph * (KHH / 32) + kc2;
            const int tap = kc >> 1;
            const int off = ((tap / 3 - 1) * PW + (tap % 3 - 1)) * NCH + (kc & 1) * 32;
            v8bf a0 = *(const v8bf*)(abase + off);
            v8bf a1 = *(const v8bf*)(abase + off + 16);
            v16bf am = join8(a0, a1);
#pragma unroll
            for (int j = 0; j < 4; ++j) {
                v16bf bm = join8(*(const v8bf*)&bpan[(j * 16 + ln) * KHH + kc2 * 32 + half * 8],
                                 *(const v8bf*)&bpan[(j * 16 + ln) * KHH + kc2 * 32 + 16 + half * 8]);
                acc[j] = __builtin_amdgcn_wmma_f32_16x16x32_bf16(
                    false, am, false, bm, (short)0, acc[j], false, false);
            }
        }
        __syncthreads();
    }
#pragma unroll
    for (int j = 0; j < 4; ++j) {
        int nc = n0 + j * 16 + ln;
        float bv = bh[nc];
#pragma unroll
        for (int v = 0; v < 8; ++v) {
            int m = m0 + v + half * 8;
            hg[(size_t)m * N3 + nc] = acc[j][v] + bv;
        }
    }
}

// ---------------------------------------------------------------------------
// Stage 5: gates + output  out[B,CH,H,W] = f*h_agg + i*tanh(...)
// ---------------------------------------------------------------------------
__global__ void gates_kernel(const float* __restrict__ xg, const float* __restrict__ hg,
                             const float* __restrict__ hagg, float* __restrict__ out) {
    int t = blockIdx.x * blockDim.x + threadIdx.x;     // NPIX*64
    if (t >= NPIX * NCH) return;
    int p = t >> 6, c = t & (NCH - 1);
    int b = p >> 14, rem = p & (HW - 1);
    size_t r = (size_t)p * N3;
    float iv = xg[r + c]       + hg[r + c];
    float fv = xg[r + 64 + c]  + hg[r + 64 + c];
    float ov = xg[r + 128 + c] + hg[r + 128 + c];
    float ig = 1.f / (1.f + expf(-iv));
    float fg = 1.f / (1.f + expf(-fv));
    float og = tanhf(ov);
    float ha = hagg[(size_t)p * NCH + c];
    out[((size_t)(b * NCH + c)) * HW + rem] = fg * ha + ig * og;
}

// ---------------------------------------------------------------------------
// Launch
// ---------------------------------------------------------------------------
static inline size_t align256(size_t v) { return (v + 255) & ~(size_t)255; }

extern "C" void kernel_launch(void* const* d_in, const int* in_sizes, int n_in,
                              void* d_out, int out_size, void* d_ws, size_t ws_size,
                              hipStream_t stream) {
    const float* x      = (const float*)d_in[0];
    const float* h_prev = (const float*)d_in[1];
    const float* Wx     = (const float*)d_in[2];
    const float* bx     = (const float*)d_in[3];
    const float* Wh     = (const float*)d_in[4];
    const float* bh     = (const float*)d_in[5];
    const float* Wf     = (const float*)d_in[6];
    const float* bfv    = (const float*)d_in[7];
    const float* Ww     = (const float*)d_in[8];
    const float* bw     = (const float*)d_in[9];
    float* out = (float*)d_out;

    const int XH_ELEMS = NB * PH * PW * CXH;     // halo-padded concat(x,h)
    const int HA_ELEMS = NB * PH * PW * NCH;     // halo-padded h_agg bf16

    char* ws = (char*)d_ws;
    size_t off = 0;
    float*  flows  = (float*) (ws + off); off = align256(off + (size_t)NPIX * 10 * 4);
    __bf16* hcat   = (__bf16*)(ws + off); off = align256(off + (size_t)NPIX * KW1 * 2);
    __bf16* haggcl = (__bf16*)(ws + off); off = align256(off + (size_t)HA_ELEMS * 2);
    float*  hagg   = (float*) (ws + off); off = align256(off + (size_t)NPIX * NCH * 4);
    float*  xg     = (float*) (ws + off); off = align256(off + (size_t)NPIX * N3 * 4);
    float*  hg     = (float*) (ws + off); off = align256(off + (size_t)NPIX * N3 * 4);
    __bf16* xh     = (__bf16*)(ws + off); off = align256(off + (size_t)XH_ELEMS * 2);
    __bf16* wtf    = (__bf16*)(ws + off); off = align256(off + (size_t)16 * KF * 2);
    __bf16* wtx    = (__bf16*)(ws + off); off = align256(off + (size_t)N3 * KX * 2);
    __bf16* wth    = (__bf16*)(ws + off); off = align256(off + (size_t)N3 * KH * 2);
    __bf16* wtw    = (__bf16*)(ws + off); off = align256(off + (size_t)NCH * KW1 * 2);

    // Repack (independent of each other)
    repack_xh<<<(XH_ELEMS + 255) / 256, 256, 0, stream>>>(x, h_prev, xh);
    fill_zero_bf16<<<(HA_ELEMS + 255) / 256, 256, 0, stream>>>(haggcl, HA_ELEMS);
    repack_wf<<<(16 * KF + 255) / 256,   256, 0, stream>>>(Wf, wtf);
    repack_wx<<<(N3 * KX + 255) / 256,   256, 0, stream>>>(Wx, wtx);
    repack_wh<<<(N3 * KH + 255) / 256,   256, 0, stream>>>(Wh, wth);
    repack_ww<<<(NCH * KW1 + 255) / 256, 256, 0, stream>>>(Ww, wtw);

    // Flow conv (WMMA) -> warp -> 1x1 aggregation (WMMA)
    gemm_flow<<<NPIX / 128, 256, 0, stream>>>(xh, wtf, bfv, flows);
    warp_kernel<<<(NB * NL * HW) / 256, 256, 0, stream>>>(h_prev, flows, hcat);
    gemm_ww<<<dim3(NPIX / 128, 1), 256, 0, stream>>>(hcat, wtw, bw, hagg, haggcl);

    // Gate convs (WMMA implicit GEMM)
    gemm_wx<<<dim3(NPIX / 128, N3 / 64), 256, 0, stream>>>(xh, wtx, bx, xg);
    gemm_wh<<<dim3(NPIX / 128, N3 / 64), 256, 0, stream>>>(haggcl, wth, bh, hg);

    // Gates + output
    gates_kernel<<<(NPIX * NCH) / 256, 256, 0, stream>>>(xg, hg, hagg, out);
}
